// AttentionCell_51591147159670
// MI455X (gfx1250) — compile-verified, matrix operand order checked by reading, not run on hardware
//
#include <hip/hip_runtime.h>
#include <math.h>

// ---------------- CDNA5 types ----------------
typedef __bf16 bf16_t;
typedef __attribute__((ext_vector_type(16))) __bf16 v16bf;
typedef __attribute__((ext_vector_type(8)))  __bf16 v8bf;
typedef __attribute__((ext_vector_type(8)))  float  v8f;
typedef __attribute__((ext_vector_type(4)))  float  v4f;

#define B_ 256
#define S_ 256
#define D_ 512
#define H_ 512
#define AS_LD 36            // fp32 LDS row stride: 36*4 = 144B, 16B-aligned rows
#define ABUF (32 * AS_LD)   // one A-tile buffer (floats)

// ---- async global -> LDS staging (gfx1250 ASYNCcnt path) ----
__device__ __forceinline__ void async_copy_b128(const float* src, float* ldsDst) {
  // generic LDS pointer: low 32 bits are the LDS byte offset
  unsigned lds = (unsigned)(unsigned long long)ldsDst;
  unsigned long long g = (unsigned long long)src;
  asm volatile("global_load_async_to_lds_b128 %0, %1, off"
               :: "v"(lds), "v"(g) : "memory");
}

template <int N>
__device__ __forceinline__ void wait_async() {
#if __has_builtin(__builtin_amdgcn_s_wait_asynccnt)
  __builtin_amdgcn_s_wait_asynccnt(N);     // N is a template constant -> ICE
#else
  asm volatile("s_wait_asynccnt %0" :: "i"(N) : "memory");
#endif
}

__device__ __forceinline__ void sched_fence() {
#if __has_builtin(__builtin_amdgcn_sched_barrier)
  __builtin_amdgcn_sched_barrier(0);   // nothing crosses: pins prefetch order
#endif
}

// ---------------------------------------------------------------------------
// Core: one block computes rows [row0, row0+32) x 512 cols of A(32xK,f32) @ B^T,
// B row-major [N x K] bf16 (GEMM column n = row n of B).
// 8 waves: wave w -> M-tile (w&1)*16, N range (w>>1)*128 (8 fragments of 16).
// A fp32 tiles are DMA'd to LDS (double-buffered async into one shared array,
// offset (i&1)*ABUF so addrspace inference keeps ds_load_b128). All 16 B-
// fragment loads issue first (sched_barrier keeps them hoisted), then the
// A-fragment ds_loads + cvts, then the 8-WMMA burst with descending loadcnt
// waits. Kernels use __launch_bounds__(256,2) so the allocator budgets ~256
// VGPRs/wave (2 waves/SIMD = one resident block) and nothing spills.
// ---------------------------------------------------------------------------
__device__ __forceinline__ void gemm_rows32(
    float* As,              // 2*ABUF floats of LDS
    const float* Ag, int lda,
    const bf16_t* Bm, int ldb, int K,
    int nbase, int am0, v8f* acc)
{
  const int t    = threadIdx.x;
  const int lane = t & 31;
  const int kh   = lane >> 4;   // which K-half this lane holds
  const int cidx = lane & 15;   // row (A) / col (B,C) within tile
  const int ar_row = t >> 3;          // 0..31
  const int ar_c4  = (t & 7) << 2;    // 0,4,...,28

  // prologue: stage k=0 tile into buffer 0
  async_copy_b128(Ag + (size_t)ar_row * lda + ar_c4,
                  As + ar_row * AS_LD + ar_c4);

  const bf16_t* brow = Bm + (size_t)(nbase + cidx) * ldb + kh * 8;
  const int nsteps = K >> 5;
#pragma unroll 1
  for (int i = 0; i < nsteps; ++i) {
    const int k0 = i << 5;

    __syncthreads();                 // all waves done reading buffer (i+1)&1
    if (i + 1 < nsteps) {
      async_copy_b128(Ag + (size_t)ar_row * lda + (k0 + 32) + ar_c4,
                      As + ((i + 1) & 1) * ABUF + ar_row * AS_LD + ar_c4);
      wait_async<1>();               // in-order: copy for step i has landed
    } else {
      wait_async<0>();
    }
    __syncthreads();

    // B prefetch: all 8 N-tiles, 16x global_load_b128 in flight
    v8bf blo[8], bhi[8];
#pragma unroll
    for (int nb = 0; nb < 8; ++nb) {
      const bf16_t* wr = brow + (size_t)(nb * 16) * ldb + k0;
      blo[nb] = *(const v8bf*)(wr);
      bhi[nb] = *(const v8bf*)(wr + 16);
    }
    sched_fence();   // keep the 16 loads hoisted ahead of ds/cvt/wmma

    // A fragment: 16x32 bf16; lane -> row cidx, K = kh*8 + {0..7, 16..23}
    // four ds_load_b128, then packed converts (B loads drain underneath)
    const float* ar = As + (i & 1) * ABUF + (am0 + cidx) * AS_LD + kh * 8;
    v4f a0 = *(const v4f*)(ar);
    v4f a1 = *(const v4f*)(ar + 4);
    v4f a2 = *(const v4f*)(ar + 16);
    v4f a3 = *(const v4f*)(ar + 20);
    v16bf afrag;
#pragma unroll
    for (int j = 0; j < 4; ++j) {
      afrag[j]      = (bf16_t)a0[j];
      afrag[j + 4]  = (bf16_t)a1[j];
      afrag[j + 8]  = (bf16_t)a2[j];
      afrag[j + 12] = (bf16_t)a3[j];
    }

#pragma unroll
    for (int nb = 0; nb < 8; ++nb) {
      v16bf bfrag = __builtin_shufflevector(blo[nb], bhi[nb],
          0,1,2,3,4,5,6,7,8,9,10,11,12,13,14,15);
      acc[nb] = __builtin_amdgcn_wmma_f32_16x16x32_bf16(
          false, afrag, false, bfrag, (short)0, acc[nb], false, false);
    }
  }
}

// ---------------- kernels ----------------
__global__ void k_f32_to_bf16(const float* __restrict__ s, bf16_t* __restrict__ d, int n) {
  int i = blockIdx.x * blockDim.x + threadIdx.x;
  if (i < n) d[i] = (bf16_t)s[i];
}

// hidden_proj = prev_h @ W_hid^T + b_hid   [256 x 512]
__global__ void __launch_bounds__(256, 2)
k_hidden_proj(const float* __restrict__ prev_h,
              const bf16_t* __restrict__ Whid,
              const float* __restrict__ b_hid,
              float* __restrict__ hp)
{
  __shared__ float As[2 * ABUF];
  const int t = threadIdx.x, w = t >> 5, lane = t & 31;
  const int am0 = (w & 1) * 16, nbase = (w >> 1) * 128;
  const int kh = lane >> 4, cidx = lane & 15;
  v8f acc[8];
  v8f z = {0,0,0,0,0,0,0,0};
#pragma unroll
  for (int nb = 0; nb < 8; ++nb) acc[nb] = z;
  const int row0 = blockIdx.x * 32;
  gemm_rows32(As, prev_h + (size_t)row0 * H_, H_, Whid, H_, H_, nbase, am0, acc);
#pragma unroll
  for (int nb = 0; nb < 8; ++nb) {
    int col = nbase + nb * 16 + cidx;
    float bias = b_hid[col];
#pragma unroll
    for (int r = 0; r < 8; ++r) {
      int row = row0 + am0 + r + 8 * kh;
      hp[(size_t)row * H_ + col] = acc[nb][r] + bias;
    }
  }
}

// energies[row] = sum_h tanh(feat_proj[row,h] + hidden_proj[b,h]) * w_score[h]
// fused: feat_proj tile never leaves registers.
__global__ void __launch_bounds__(256, 2)
k_energies(const float* __restrict__ batch_H,
           const bf16_t* __restrict__ Wfeat,
           const float* __restrict__ hidden_proj,
           const float* __restrict__ w_score,
           float* __restrict__ energies)
{
  __shared__ float As[2 * ABUF];
  __shared__ float hp[H_];
  __shared__ float wsc[H_];
  __shared__ float en[32];
  const int t = threadIdx.x, w = t >> 5, lane = t & 31;
  const int am0 = (w & 1) * 16, nbase = (w >> 1) * 128;
  const int kh = lane >> 4, cidx = lane & 15;
  const int row0 = blockIdx.x * 32;      // 32 consecutive (b,s) rows, same b
  const int b = row0 >> 8;               // row0 / S
  for (int i = t; i < H_; i += 256) {
    hp[i]  = hidden_proj[(size_t)b * H_ + i];
    wsc[i] = w_score[i];
  }
  if (t < 32) en[t] = 0.f;
  v8f acc[8];
  v8f z = {0,0,0,0,0,0,0,0};
#pragma unroll
  for (int nb = 0; nb < 8; ++nb) acc[nb] = z;
  gemm_rows32(As, batch_H + (size_t)row0 * D_, D_, Wfeat, D_, D_, nbase, am0, acc);

  float es[8] = {0.f,0.f,0.f,0.f,0.f,0.f,0.f,0.f};
#pragma unroll
  for (int nb = 0; nb < 8; ++nb) {
    int col = nbase + nb * 16 + cidx;
    float wv = wsc[col], hb = hp[col];
#pragma unroll
    for (int r = 0; r < 8; ++r) es[r] += tanhf(acc[nb][r] + hb) * wv;
  }
#pragma unroll
  for (int r = 0; r < 8; ++r) {
    float v = es[r];
#pragma unroll
    for (int off = 1; off < 16; off <<= 1) v += __shfl_xor(v, off, 16);
    if (cidx == 0) atomicAdd(&en[am0 + r + 8 * kh], v);
  }
  __syncthreads();
  if (t < 32) energies[row0 + t] = en[t];
}

__global__ void k_softmax(const float* __restrict__ energies,
                          float* __restrict__ alpha_ws,
                          float* __restrict__ alpha_out)
{
  __shared__ float red[256];
  int b = blockIdx.x, t = threadIdx.x;
  float e = energies[b * 256 + t];
  red[t] = e; __syncthreads();
  for (int s = 128; s > 0; s >>= 1) { if (t < s) red[t] = fmaxf(red[t], red[t + s]); __syncthreads(); }
  float m = red[0]; __syncthreads();
  float ex = expf(e - m);
  red[t] = ex; __syncthreads();
  for (int s = 128; s > 0; s >>= 1) { if (t < s) red[t] += red[t + s]; __syncthreads(); }
  float a = ex / red[0];
  alpha_ws[b * 256 + t]  = a;
  alpha_out[b * 256 + t] = a;
}

// context[b,:] = sum_s alpha[b,s] * batch_H[b,s,:]   (second pass over batch_H)
__global__ void k_context(const float* __restrict__ batch_H,
                          const float* __restrict__ alpha,
                          float* __restrict__ context)
{
  __shared__ float al[S_];
  int b = blockIdx.x, t = threadIdx.x;
  al[t] = alpha[b * 256 + t];
  __syncthreads();
  float c0 = 0.f, c1 = 0.f;
  const float* base = batch_H + (size_t)b * S_ * D_;
  for (int s = 0; s < S_; ++s) {
    if ((s & 7) == 0 && s + 16 < S_) {
      // stream-ahead hint: pulls the cacheline 16 rows ahead (global_prefetch)
      __builtin_prefetch(base + (size_t)(s + 16) * D_ + t, 0, 1);
    }
    float a = al[s];
    c0 = fmaf(a, base[(size_t)s * D_ + t], c0);
    c1 = fmaf(a, base[(size_t)s * D_ + t + 256], c1);
  }
  context[b * D_ + t]       = c0;
  context[b * D_ + t + 256] = c1;
}

__global__ void k_build_x(const float* __restrict__ context,
                          const float* __restrict__ onehot,
                          float* __restrict__ x)
{
  int i = blockIdx.x * 256 + threadIdx.x;   // < B*1024
  int b = i >> 10, j = i & 1023;
  x[i] = (j < 512) ? context[b * 512 + j] : onehot[b * 512 + (j - 512)];
}

// gates = x @ W_ih^T + prev_h @ W_hh^T + b_ih + b_hh   [256 x 2048]
__global__ void __launch_bounds__(256, 2)
k_gates(const float* __restrict__ x,
        const float* __restrict__ prev_h,
        const bf16_t* __restrict__ Wih,
        const bf16_t* __restrict__ Whh,
        const float* __restrict__ b_ih,
        const float* __restrict__ b_hh,
        float* __restrict__ gates)
{
  __shared__ float As[2 * ABUF];
  const int t = threadIdx.x, w = t >> 5, lane = t & 31;
  const int am0 = (w & 1) * 16, nbase = (w >> 1) * 128;
  const int kh = lane >> 4, cidx = lane & 15;
  const int row0  = blockIdx.x * 32;
  const int jbase = blockIdx.y * 512;
  v8f acc[8];
  v8f z = {0,0,0,0,0,0,0,0};
#pragma unroll
  for (int nb = 0; nb < 8; ++nb) acc[nb] = z;
  gemm_rows32(As, x + (size_t)row0 * 1024, 1024,
              Wih + (size_t)jbase * 1024, 1024, 1024, nbase, am0, acc);
  gemm_rows32(As, prev_h + (size_t)row0 * H_, H_,
              Whh + (size_t)jbase * H_, H_, H_, nbase, am0, acc);
#pragma unroll
  for (int nb = 0; nb < 8; ++nb) {
    int j = jbase + nbase + nb * 16 + cidx;
    float bias = b_ih[j] + b_hh[j];
#pragma unroll
    for (int r = 0; r < 8; ++r) {
      int row = row0 + am0 + r + 8 * kh;
      gates[(size_t)row * 2048 + j] = acc[nb][r] + bias;
    }
  }
}

__global__ void k_lstm(const float* __restrict__ gates,
                       const float* __restrict__ prev_c,
                       float* __restrict__ h_new,
                       float* __restrict__ c_new)
{
  int i = blockIdx.x * 256 + threadIdx.x;   // < B*H
  int b = i >> 9, h = i & 511;
  const float* g = gates + (size_t)b * 2048;
  float gi = g[h], gf = g[512 + h], gg = g[1024 + h], go = g[1536 + h];
  float si = 1.f / (1.f + expf(-gi));
  float sf = 1.f / (1.f + expf(-gf));
  float so = 1.f / (1.f + expf(-go));
  float c  = sf * prev_c[i] + si * tanhf(gg);
  h_new[i] = so * tanhf(c);
  c_new[i] = c;
}

// ---------------- launch ----------------
extern "C" void kernel_launch(void* const* d_in, const int* in_sizes, int n_in,
                              void* d_out, int out_size, void* d_ws, size_t ws_size,
                              hipStream_t stream) {
  (void)in_sizes; (void)n_in; (void)out_size; (void)ws_size;
  const float* prev_h  = (const float*)d_in[0];
  const float* prev_c  = (const float*)d_in[1];
  const float* batch_H = (const float*)d_in[2];
  const float* onehot  = (const float*)d_in[3];
  const float* W_feat  = (const float*)d_in[4];
  const float* W_hid   = (const float*)d_in[5];
  const float* b_hid   = (const float*)d_in[6];
  const float* w_score = (const float*)d_in[7];
  const float* W_ih    = (const float*)d_in[8];
  const float* W_hh    = (const float*)d_in[9];
  const float* b_ih    = (const float*)d_in[10];
  const float* b_hh    = (const float*)d_in[11];

  char* wsb = (char*)d_ws;                         // ~11.5 MB used
  bf16_t* Wfeat_bf   = (bf16_t*)(wsb + 0);         // 512 KB
  bf16_t* Whid_bf    = (bf16_t*)(wsb + 524288);    // 512 KB
  bf16_t* Wih_bf     = (bf16_t*)(wsb + 1048576);   // 4 MB
  bf16_t* Whh_bf     = (bf16_t*)(wsb + 5242880);   // 2 MB
  float* hidden_proj = (float*)(wsb + 7340032);    // 512 KB
  float* energies    = (float*)(wsb + 7864320);    // 256 KB
  float* alpha_ws    = (float*)(wsb + 8126464);    // 256 KB
  float* context     = (float*)(wsb + 8388608);    // 512 KB
  float* x           = (float*)(wsb + 8912896);    // 1 MB
  float* gates       = (float*)(wsb + 9961472);    // 2 MB

  float* out       = (float*)d_out;
  float* h_new     = out;
  float* c_new     = out + 256 * 512;
  float* alpha_out = out + 2 * 256 * 512;

  k_f32_to_bf16<<<1024, 256, 0, stream>>>(W_feat, Wfeat_bf, 262144);
  k_f32_to_bf16<<<1024, 256, 0, stream>>>(W_hid,  Whid_bf,  262144);
  k_f32_to_bf16<<<8192, 256, 0, stream>>>(W_ih,   Wih_bf,   2097152);
  k_f32_to_bf16<<<4096, 256, 0, stream>>>(W_hh,   Whh_bf,   1048576);

  k_hidden_proj<<<8,    256, 0, stream>>>(prev_h, Whid_bf, b_hid, hidden_proj);
  k_energies  <<<2048, 256, 0, stream>>>(batch_H, Wfeat_bf, hidden_proj, w_score, energies);
  k_softmax   <<<256,  256, 0, stream>>>(energies, alpha_ws, alpha_out);
  k_context   <<<256,  256, 0, stream>>>(batch_H, alpha_ws, context);
  k_build_x   <<<1024, 256, 0, stream>>>(context, onehot, x);
  k_gates     <<<dim3(8, 4), 256, 0, stream>>>(x, prev_h, Wih_bf, Whh_bf, b_ih, b_hh, gates);
  k_lstm      <<<512,  256, 0, stream>>>(gates, prev_c, h_new, c_new);
}
